// UniAttention_18554258719245
// MI455X (gfx1250) — matrix operand order repaired and checked by reading, not compiled, not run on hardware
//
#include <hip/hip_runtime.h>
#include <hip/hip_bf16.h>
#include <cstdint>
#include <cmath>

typedef unsigned short u16;
typedef unsigned int   u32;

typedef __attribute__((ext_vector_type(16))) __bf16 v16bf;
typedef __attribute__((ext_vector_type(8)))  float  v8f;

union Frag16 {           // 16 bf16 = 32 bytes = two uint4
    v16bf v;
    uint4 q[2];
};

__device__ __forceinline__ u16 f32_to_bf16(float f) {
    union { float f; u32 u; } x{f};
    u32 r = x.u + 0x7FFFu + ((x.u >> 16) & 1u);   // round-to-nearest-even
    return (u16)(r >> 16);
}

// ---- CDNA5 async copy global->LDS (ASYNCcnt-tracked DMA, no VGPR round trip)
__device__ __forceinline__ void async_ld_b128(u32 lds_addr, const void* gaddr) {
    asm volatile("global_load_async_to_lds_b128 %0, %1, off"
                 :: "v"(lds_addr), "v"(gaddr)
                 : "memory");
}
#define ASYNC_WAIT(N) asm volatile("s_wait_asynccnt " #N ::: "memory")

// LDS byte offset of a __shared__ object (generic addr low 32 bits == LDS offset)
__device__ __forceinline__ u32 lds_off(const void* p) {
    return (u32)(size_t)p;
}

// ---------------------------------------------------------------------------
// Prep: xo = bf16(x), xp = bf16(x + pos)  (pos broadcast over batch)
// ---------------------------------------------------------------------------
__global__ __launch_bounds__(256)
void prep_x_kernel(const float* __restrict__ x, const float* __restrict__ pos,
                   u16* __restrict__ xp, u16* __restrict__ xo, long long SH, long long total) {
    long long i = (long long)blockIdx.x * blockDim.x + threadIdx.x;
    if (i >= total) return;
    long long sh = i % SH;
    float xv = x[i];
    xo[i] = f32_to_bf16(xv);
    xp[i] = f32_to_bf16(xv + pos[sh]);
}

// ---------------------------------------------------------------------------
// Weight transpose + f32->bf16: W[K][N] -> Wt[N][K]
// ---------------------------------------------------------------------------
__global__ __launch_bounds__(256)
void transpose_bf16_kernel(const float* __restrict__ W, u16* __restrict__ Wt,
                           int K, int N) {
    __shared__ float tile[32][33];
    int n0 = blockIdx.x * 32;
    int k0 = blockIdx.y * 32;
    int tx = threadIdx.x & 31;
    int ty = threadIdx.x >> 5;          // 0..7
#pragma unroll
    for (int r = ty; r < 32; r += 8)
        tile[r][tx] = W[(size_t)(k0 + r) * N + n0 + tx];
    __syncthreads();
#pragma unroll
    for (int r = ty; r < 32; r += 8)
        Wt[(size_t)(n0 + r) * K + k0 + tx] = f32_to_bf16(tile[tx][r]);
}

// ---------------------------------------------------------------------------
// Tiled bf16 GEMM via v_wmma_f32_16x16x32_bf16 with double-buffered LDS and
// async global->LDS DMA (stage k+1 overlaps WMMAs of stage k).
//   C[M][N] = A[M][K] * Bt[N][K]^T   (A,Bt bf16 row-major; Bt holds B^T)
// Block 256 thr = 8 waves, tile 128x128x32. Wave (wm in 0..3, wn in 0..1)
// computes 32x64 = 2x4 accumulators of 16x16.
// ---------------------------------------------------------------------------
#define G_BM 128
#define G_BN 128
#define G_BK 32
#define G_LDA 40    // padded LDS stride (bf16 elems) for a 32-wide K tile

template <bool OUT_BF16>
__global__ __launch_bounds__(256)
void gemm_bf16_wmma(const u16* __restrict__ A, const u16* __restrict__ Bt,
                    void* __restrict__ Cout, int M, int N, int K) {
    __shared__ u16 As[2][G_BM * G_LDA];
    __shared__ u16 Bs[2][G_BN * G_LDA];

    const int tid  = threadIdx.x;
    const int lane = tid & 31;
    const int wave = tid >> 5;
    const int wm   = wave & 3;          // 4 waves along M
    const int wn   = wave >> 2;         // 2 waves along N
    const int lrow = lane & 15;
    const int half = lane >> 4;

    const int bm = blockIdx.x * G_BM;
    const int bn = blockIdx.y * G_BN;

    const u32 as0 = lds_off(&As[0][0]);
    const u32 bs0 = lds_off(&Bs[0][0]);
    const u32 stg = (u32)(G_BM * G_LDA * 2);   // stage size in bytes

    v8f acc[2][4] = {};

    // per-stage async issue: 4 x b128 per thread (2 A + 2 B)
    auto issue_stage = [&](int k0, int buf) {
#pragma unroll
        for (int i = 0; i < 2; ++i) {
            int idx = tid + i * 256;           // 0..511
            int row = idx >> 2;
            int c8  = (idx & 3) * 8;
            u32 loff = (u32)((row * G_LDA + c8) * 2) + (u32)buf * stg;
            async_ld_b128(as0 + loff, A  + (size_t)(bm + row) * K + k0 + c8);
            async_ld_b128(bs0 + loff, Bt + (size_t)(bn + row) * K + k0 + c8);
        }
    };

    const int KT = K / G_BK;
    issue_stage(0, 0);

    for (int kt = 0; kt < KT; ++kt) {
        const int buf = kt & 1;
        __syncthreads();                       // buffer (kt+1)&1 free for DMA
        if (kt + 1 < KT) {
            issue_stage((kt + 1) * G_BK, (kt + 1) & 1);
            ASYNC_WAIT(4);                     // drain stage kt, keep kt+1 in flight
        } else {
            ASYNC_WAIT(0);
        }
        __syncthreads();                       // stage kt visible to all waves

        // --- fragments ---
        Frag16 fa[2], fb[4];
#pragma unroll
        for (int t = 0; t < 2; ++t) {     // A: lanes<16 get K {0..7,16..23}
            const u16* p = &As[buf][(wm * 32 + t * 16 + lrow) * G_LDA + half * 8];
            fa[t].q[0] = *(const uint4*)(p);
            fa[t].q[1] = *(const uint4*)(p + 16);
        }
#pragma unroll
        for (int t = 0; t < 4; ++t) {     // B: lanes<16 get K {0..7,16..23}
            const u16* p = &Bs[buf][(wn * 64 + t * 16 + lrow) * G_LDA + half * 8];
            fb[t].q[0] = *(const uint4*)(p);
            fb[t].q[1] = *(const uint4*)(p + 16);
        }
#pragma unroll
        for (int i = 0; i < 2; ++i)
#pragma unroll
            for (int j = 0; j < 4; ++j)
                acc[i][j] = __builtin_amdgcn_wmma_f32_16x16x32_bf16(
                    false, fa[i].v, false, fb[j].v, (short)0, acc[i][j], false, false);
    }

    // --- store: C element (M = r + 8*half, N = lrow) per 16x16 tile ---
#pragma unroll
    for (int i = 0; i < 2; ++i) {
#pragma unroll
        for (int j = 0; j < 4; ++j) {
#pragma unroll
            for (int r = 0; r < 8; ++r) {
                int row = bm + wm * 32 + i * 16 + 8 * half + r;
                int col = bn + wn * 64 + j * 16 + lrow;
                if (OUT_BF16)
                    ((u16*)Cout)[(size_t)row * N + col] = f32_to_bf16(acc[i][j][r]);
                else
                    ((float*)Cout)[(size_t)row * N + col] = acc[i][j][r];
            }
        }
    }
}

// ---------------------------------------------------------------------------
// Flash attention (causal). Block = 128 thr (4 waves), 64 q-rows per block
// (16 per wave). Iterates key blocks of 64. hs fixed at 64.
//   qk: [B*S][2H] bf16 (Q cols [0,H), K cols [H,2H)),  v: [B*S][H] bf16
//   o : [B*S][H] bf16
// K tile staged with async DMA; V tile loaded to VGPRs and stored transposed.
// ---------------------------------------------------------------------------
#define A_LDS 72     // bf16 stride for 64-wide tiles
#define S_LDS 66     // f32 stride for score tiles

__global__ __launch_bounds__(128)
void attention_kernel(const u16* __restrict__ qk, const u16* __restrict__ v,
                      u16* __restrict__ o, int B, int S, int H, int nh) {
    const int hs = H / nh;                 // 64
    const float scale = rsqrtf((float)hs); // 0.125

    __shared__ u16   Ks[64 * A_LDS];       // [key][d]
    __shared__ u16   Vt[64 * A_LDS];       // [d][key]  (transposed)
    __shared__ float Ss[4][16 * S_LDS];    // per-wave raw scores (f32)
    __shared__ u16   Ps[4][16 * A_LDS];    // per-wave probs (bf16)
    __shared__ float rowm[4][16], rowl[4][16], rowa[4][16];

    const int tid  = threadIdx.x;
    const int lane = tid & 31;
    const int wave = tid >> 5;             // 0..3
    const int lrow = lane & 15;
    const int half = lane >> 4;

    const int qb = blockIdx.x;             // S/64 blocks
    const int h  = blockIdx.y;             // head
    const int b  = blockIdx.z;             // batch
    const int H2 = 2 * H;

    const u32 ks_base = lds_off(&Ks[0]);

    if (lane < 16) {
        rowm[wave][lane] = -3.0e38f;
        rowl[wave][lane] = 0.0f;
        rowa[wave][lane] = 1.0f;
    }

    // --- Q fragments for this wave's 16 rows (held for the whole kernel) ---
    const u16* qbase = qk + ((size_t)b * S + (size_t)qb * 64 + wave * 16 + lrow) * H2 + h * hs;
    Frag16 qa[2];
#pragma unroll
    for (int t = 0; t < 2; ++t) {          // d-chunks 0..31, 32..63
        const u16* p = qbase + t * 32 + half * 8;
        qa[t].q[0] = *(const uint4*)(p);
        qa[t].q[1] = *(const uint4*)(p + 16);
    }

    v8f oacc[4] = {};

    const u16* kbaseg = qk + ((size_t)b * S) * H2 + H + h * hs;  // K columns
    const u16* vbaseg = v  + ((size_t)b * S) * H  + h * hs;

    for (int kb = 0; kb <= qb; ++kb) {
        // ---- K block: async DMA straight into LDS (key-major) ----
#pragma unroll
        for (int i = 0; i < 4; ++i) {
            int idx = tid + i * 128;                // 0..511
            int row = idx >> 3;                     // key local 0..63
            int c8  = (idx & 7) * 8;                // d offset
            async_ld_b128(ks_base + (u32)((row * A_LDS + c8) * 2),
                          kbaseg + ((size_t)kb * 64 + row) * H2 + c8);
        }
        // ---- V block: batch 4 global loads, then transpose-store to LDS ----
        uint4 vq[4];
#pragma unroll
        for (int i = 0; i < 4; ++i) {
            int idx = tid + i * 128;
            int row = idx >> 3;                     // key local
            int c8  = (idx & 7) * 8;                // d base
            vq[i] = *(const uint4*)(vbaseg + ((size_t)kb * 64 + row) * H + c8);
        }
#pragma unroll
        for (int i = 0; i < 4; ++i) {
            int idx = tid + i * 128;
            int row = idx >> 3;
            int c8  = (idx & 7) * 8;
            const u16* e = (const u16*)&vq[i];
#pragma unroll
            for (int j = 0; j < 8; ++j)
                Vt[(c8 + j) * A_LDS + row] = e[j];
        }
        ASYNC_WAIT(0);                              // K DMA complete
        __syncthreads();

        // ---- S = Q K^T : 4 key-tiles x 2 d-steps of WMMA ----
        v8f sacc[4] = {};
#pragma unroll
        for (int j = 0; j < 4; ++j) {
#pragma unroll
            for (int t = 0; t < 2; ++t) {
                Frag16 fb;
                const u16* p = Ks + (j * 16 + lrow) * A_LDS + t * 32 + half * 8;
                fb.q[0] = *(const uint4*)(p);
                fb.q[1] = *(const uint4*)(p + 16);
                sacc[j] = __builtin_amdgcn_wmma_f32_16x16x32_bf16(
                    false, qa[t].v, false, fb.v, (short)0, sacc[j], false, false);
            }
        }
#pragma unroll
        for (int j = 0; j < 4; ++j)
#pragma unroll
            for (int r = 0; r < 8; ++r)
                Ss[wave][(r + 8 * half) * S_LDS + j * 16 + lrow] = sacc[j][r];
        __syncthreads();

        // ---- online softmax: 2 lanes per row, 32 cols each ----
        {
            int srow  = lane >> 1;
            int shalf = (lane & 1) * 32;
            int qglob = qb * 64 + wave * 16 + srow;
            int kcol0 = kb * 64 + shalf;
            const float* sp = &Ss[wave][srow * S_LDS + shalf];

            float mx = -3.0e38f;
#pragma unroll
            for (int j = 0; j < 32; ++j) {
                float s = sp[j] * scale;
                if (kcol0 + j > qglob) s = -10000.0f;
                mx = fmaxf(mx, s);
            }
            mx = fmaxf(mx, __shfl_xor(mx, 1, 32));

            float mold = rowm[wave][srow];
            float lold = rowl[wave][srow];
            float mnew = fmaxf(mold, mx);
            float alpha = __expf(mold - mnew);

            float sum = 0.0f;
            u16* pp = &Ps[wave][srow * A_LDS + shalf];
#pragma unroll
            for (int j = 0; j < 32; ++j) {
                float s = sp[j] * scale;
                if (kcol0 + j > qglob) s = -10000.0f;
                float p = __expf(s - mnew);
                sum += p;
                pp[j] = f32_to_bf16(p);
            }
            sum += __shfl_xor(sum, 1, 32);
            if ((lane & 1) == 0) {
                rowm[wave][srow] = mnew;
                rowl[wave][srow] = alpha * lold + sum;
                rowa[wave][srow] = alpha;
            }
        }
        __syncthreads();

        // ---- rescale running O by alpha, then O += P @ V ----
        float av[8];
#pragma unroll
        for (int r = 0; r < 8; ++r) av[r] = rowa[wave][8 * half + r];
#pragma unroll
        for (int j = 0; j < 4; ++j)
#pragma unroll
            for (int r = 0; r < 8; ++r) oacc[j][r] *= av[r];

        Frag16 pa[2];
#pragma unroll
        for (int t = 0; t < 2; ++t) {     // keys 0..31, 32..63 as A-frag
            const u16* p = &Ps[wave][lrow * A_LDS + t * 32 + half * 8];
            pa[t].q[0] = *(const uint4*)(p);
            pa[t].q[1] = *(const uint4*)(p + 16);
        }
#pragma unroll
        for (int j = 0; j < 4; ++j) {     // d-tiles
#pragma unroll
            for (int t = 0; t < 2; ++t) {
                Frag16 fb;
                const u16* p = Vt + (j * 16 + lrow) * A_LDS + t * 32 + half * 8;
                fb.q[0] = *(const uint4*)(p);
                fb.q[1] = *(const uint4*)(p + 16);
                oacc[j] = __builtin_amdgcn_wmma_f32_16x16x32_bf16(
                    false, pa[t].v, false, fb.v, (short)0, oacc[j], false, false);
            }
        }
        __syncthreads();
    }

    // ---- finalize: divide by row sums, write bf16 ----
    float linv[8];
#pragma unroll
    for (int r = 0; r < 8; ++r) linv[r] = 1.0f / rowl[wave][8 * half + r];

    u16* obase = o + ((size_t)b * S + (size_t)qb * 64 + wave * 16) * H + h * hs;
#pragma unroll
    for (int j = 0; j < 4; ++j)
#pragma unroll
        for (int r = 0; r < 8; ++r) {
            int row = 8 * half + r;
            obase[(size_t)row * H + j * 16 + lrow] = f32_to_bf16(oacc[j][r] * linv[r]);
        }
}

// ---------------------------------------------------------------------------
// Host launcher
// ---------------------------------------------------------------------------
extern "C" void kernel_launch(void* const* d_in, const int* in_sizes, int n_in,
                              void* d_out, int out_size, void* d_ws, size_t ws_size,
                              hipStream_t stream) {
    (void)n_in; (void)out_size; (void)ws_size;
    const float* x    = (const float*)d_in[0];
    const float* pos  = (const float*)d_in[1];
    const float* Wqkv = (const float*)d_in[2];
    const float* Wout = (const float*)d_in[3];

    const int H  = (int)llround(sqrt((double)in_sizes[3]));  // W_out is H x H
    const int S  = in_sizes[1] / H;                          // pos_table is S x H
    const int B  = (int)((long long)in_sizes[0] / ((long long)S * H));
    const int BS = B * S;
    const int nh = 16;

    char* ws = (char*)d_ws;
    size_t off = 0;
    auto alloc = [&](size_t bytes) -> char* {
        char* p = ws + off;
        off += (bytes + 255) & ~(size_t)255;
        return p;
    };
    u16* xp    = (u16*)alloc((size_t)BS * H * 2);        // bf16(x+pos)
    u16* xb    = (u16*)alloc((size_t)BS * H * 2);        // bf16(x)
    u16* Wqkvt = (u16*)alloc((size_t)3 * H * H * 2);     // [3H][H]
    u16* Woutt = (u16*)alloc((size_t)H * H * 2);         // [H][H]
    u16* qkp   = (u16*)alloc((size_t)BS * 2 * H * 2);    // [BS][2H]
    u16* vv    = (u16*)alloc((size_t)BS * H * 2);        // [BS][H]
    u16* attn  = (u16*)alloc((size_t)BS * H * 2);        // [BS][H]

    // 1) activations -> bf16 (+pos)
    {
        long long total = (long long)BS * H;
        long long SH = (long long)S * H;
        int blocks = (int)((total + 255) / 256);
        prep_x_kernel<<<blocks, 256, 0, stream>>>(x, pos, xp, xb, SH, total);
    }
    // 2) weight transposes -> bf16
    transpose_bf16_kernel<<<dim3(3 * H / 32, H / 32), 256, 0, stream>>>(Wqkv, Wqkvt, H, 3 * H);
    transpose_bf16_kernel<<<dim3(H / 32, H / 32), 256, 0, stream>>>(Wout, Woutt, H, H);

    // 3) QK projection: (x+pos) @ Wqkv[:, :2H]   -> qkp [BS][2H] bf16
    gemm_bf16_wmma<true><<<dim3(BS / G_BM, 2 * H / G_BN), 256, 0, stream>>>(
        xp, Wqkvt, qkp, BS, 2 * H, H);
    // 4) V projection: x @ Wqkv[:, 2H:]          -> vv [BS][H] bf16
    gemm_bf16_wmma<true><<<dim3(BS / G_BM, H / G_BN), 256, 0, stream>>>(
        xb, Wqkvt + (size_t)2 * H * H, vv, BS, H, H);

    // 5) causal flash attention -> attn [BS][H] bf16
    attention_kernel<<<dim3(S / 64, nh, B), 128, 0, stream>>>(qkp, vv, attn, B, S, H, nh);

    // 6) output projection: attn @ W_out -> d_out (f32)
    gemm_bf16_wmma<false><<<dim3(BS / G_BM, H / G_BN), 256, 0, stream>>>(
        attn, Woutt, (float*)d_out, BS, H, H);
}
